// BaseDecoder_45062796869667
// MI455X (gfx1250) — matrix operand order repaired and checked by reading, not compiled, not run on hardware
//
#include <hip/hip_runtime.h>
#include <hip/hip_bf16.h>
#include <math.h>

// ---------------------------------------------------------------------------
// BaseDecoder forward for MI455X (gfx1250, wave32).
// - All matmuls via v_wmma_f32_16x16x32_bf16 (bf16 in, f32 accum).
// - GEMM: 128x128 block tile, 8 waves x (32x64) sub-tiles, double-buffered
//   LDS software pipeline + global_prefetch two K-steps ahead.
// - Attention bias fused into score epilogue (no [B,H,S,S] bias tensor).
// - Softmax keeps the row in registers and emits bf16 probabilities, which
//   attn_av consumes directly as WMMA A-fragments.
// ---------------------------------------------------------------------------

#define B_   32
#define S_   256
#define M_   128
#define E_   1024
#define H_   16
#define DH_  64
#define F_   4096
#define L_   6
#define V_   200

typedef __attribute__((ext_vector_type(16))) __bf16 v16bf;
typedef __attribute__((ext_vector_type(8)))  float  v8f;

// ---- WMMA fragment loaders (contiguous k order, same for A and B) -----
// A (16x32, MxK): lanes 0-15 row m; elem e -> K = e + (lane>=16?16:0)
__device__ __forceinline__ v16bf frag_a_lds(const __bf16* base, int ldk, int lane) {
    v16bf a;
    const __bf16* row = base + (size_t)(lane & 15) * ldk;
    const int koff = (lane & 16) ? 16 : 0;
#pragma unroll
    for (int e = 0; e < 16; ++e) {
        int k = e + koff;
        a[e] = row[k];
    }
    return a;
}
// B (32x16, KxN): lane = col n; elem e -> K = e + (lane>=16?16:0); K contiguous
__device__ __forceinline__ v16bf frag_b_lds(const __bf16* base, int ldk, int lane) {
    v16bf b;
    const __bf16* row = base + (size_t)(lane & 15) * ldk;
    const int koff = (lane & 16) ? 16 : 0;
#pragma unroll
    for (int e = 0; e < 16; ++e) b[e] = row[e + koff];
    return b;
}

// ---------------------------------------------------------------------------
// Generic GEMM: C[M,N] = act(A[M,K] @ W[N,K]^T + bias[N])
// A,W f32 row-major, staged to LDS as bf16, double-buffered pipeline.
// M % 128 == 0, K % 32 == 0. act: 0 = none, 1 = exact GELU.
// ---------------------------------------------------------------------------
#define TM 128
#define TN 128
#define TK 32
__global__ __launch_bounds__(256)
void gemm_bf16_kernel(const float* __restrict__ A, const float* __restrict__ W,
                      const float* __restrict__ bias, float* __restrict__ C,
                      int Mtot, int N, int K, int act)
{
    __shared__ __bf16 As[2][TM * TK];
    __shared__ __bf16 Bs[2][TN * TK];
    const int tid  = threadIdx.x;
    const int lane = tid & 31;
    const int wave = tid >> 5;
    const int row0 = blockIdx.y * TM;
    const int col0 = blockIdx.x * TN;
    const int wr = (wave >> 1) * 32;   // 0..96  (M direction, 4 waves)
    const int wc = (wave & 1) * 64;    // 0/64   (N direction, 2 waves)

    // Per-thread staging: 16 f32 of A and 16 f32 of W per K-step.
    const int lr = tid >> 1;             // 0..127 (row of A tile / row of W tile)
    const int lc = (tid & 1) * 16;       // 0 / 16
    const float* aptr = A + (size_t)(row0 + lr) * K + lc;
    const float* bptr = W + (size_t)(col0 + lr) * K + lc;
    const bool bvalid = (col0 + lr) < N;

    float4 areg[4], breg[4];
#pragma unroll
    for (int i = 0; i < 4; ++i) breg[i] = (float4){0.f, 0.f, 0.f, 0.f};

    auto gload = [&](int k0) {
        const float4* ap = (const float4*)(aptr + k0);
#pragma unroll
        for (int i = 0; i < 4; ++i) areg[i] = ap[i];
        if (bvalid) {
            const float4* bp = (const float4*)(bptr + k0);
#pragma unroll
            for (int i = 0; i < 4; ++i) breg[i] = bp[i];
        }
        if (k0 + 2 * TK < K) {           // prefetch two K-steps ahead
            __builtin_prefetch(aptr + k0 + 2 * TK, 0, 0);
            if (bvalid) __builtin_prefetch(bptr + k0 + 2 * TK, 0, 0);
        }
    };
    auto lstore = [&](int buf) {
        __bf16* da = As[buf] + lr * TK + lc;
        __bf16* db = Bs[buf] + lr * TK + lc;
#pragma unroll
        for (int i = 0; i < 4; ++i) {
            da[4 * i + 0] = (__bf16)areg[i].x; da[4 * i + 1] = (__bf16)areg[i].y;
            da[4 * i + 2] = (__bf16)areg[i].z; da[4 * i + 3] = (__bf16)areg[i].w;
            db[4 * i + 0] = (__bf16)breg[i].x; db[4 * i + 1] = (__bf16)breg[i].y;
            db[4 * i + 2] = (__bf16)breg[i].z; db[4 * i + 3] = (__bf16)breg[i].w;
        }
    };

    v8f acc[2][4];
#pragma unroll
    for (int i = 0; i < 2; ++i)
#pragma unroll
        for (int j = 0; j < 4; ++j) acc[i][j] = (v8f){0,0,0,0,0,0,0,0};

    gload(0);
    lstore(0);
    __syncthreads();
    int buf = 0;
    for (int k0 = 0; k0 < K; k0 += TK) {
        const bool has_next = (k0 + TK) < K;
        if (has_next) gload(k0 + TK);          // overlap with wmma below
        v16bf a[2];
#pragma unroll
        for (int i = 0; i < 2; ++i)
            a[i] = frag_a_lds(As[buf] + (wr + 16 * i) * TK, TK, lane);
#pragma unroll
        for (int j = 0; j < 4; ++j) {
            v16bf b = frag_b_lds(Bs[buf] + (wc + 16 * j) * TK, TK, lane);
#pragma unroll
            for (int i = 0; i < 2; ++i)
                acc[i][j] = __builtin_amdgcn_wmma_f32_16x16x32_bf16(
                    false, a[i], false, b, (short)0, acc[i][j], false, false);
        }
        if (has_next) {
            lstore(buf ^ 1);                   // other buffer: no pre-store barrier
            __syncthreads();
            buf ^= 1;
        }
    }

    const int nbase = col0 + wc + (lane & 15);
    const int mbase = row0 + wr + ((lane & 16) ? 8 : 0);
#pragma unroll
    for (int i = 0; i < 2; ++i)
#pragma unroll
        for (int j = 0; j < 4; ++j) {
            int col = nbase + 16 * j;
            if (col >= N) continue;
            float bv = bias ? bias[col] : 0.0f;
#pragma unroll
            for (int v = 0; v < 8; ++v) {
                int row = mbase + 16 * i + v;
                float val = acc[i][j][v] + bv;
                if (act == 1)
                    val = 0.5f * val * (1.0f + erff(val * 0.70710678118654752f));
                C[(size_t)row * N + col] = val;
            }
        }
}

// ---------------------------------------------------------------------------
// Attention scores: S[z, sq, sk] = (Q_z[sq,:] . K_z[sk,:]) / 8 + bias(...)
// z = b*H + h. One wave per 32x32 tile; WMMA over DH=64 (two K-steps).
// mode 0 (self): bias = dist_emb + iso_emb, causal + pad mask (fused).
// mode 1 (cross): bias = memmask ? -inf : 0.
// ---------------------------------------------------------------------------
__global__ __launch_bounds__(32)
void attn_scores_kernel(const float* __restrict__ Q, int q_stride,
                        const float* __restrict__ Km, int k_stride,
                        float* __restrict__ Sout, int Sq, int Sk,
                        const int* __restrict__ dist, const int* __restrict__ iso,
                        const float* __restrict__ dist_w, const float* __restrict__ iso_w,
                        const int* __restrict__ seqs,
                        const unsigned char* __restrict__ memmask, int mode)
{
    const int lane = threadIdx.x & 31;
    const int z = blockIdx.z;
    const int b = z / H_, h = z % H_;
    const int qtile = blockIdx.y * 32;
    const int ktile = blockIdx.x * 32;
    const float* Qb = Q  + (size_t)b * Sq * q_stride + h * DH_;
    const float* Kb = Km + (size_t)b * Sk * k_stride + h * DH_;

    v8f acc[2][2];
#pragma unroll
    for (int i = 0; i < 2; ++i)
#pragma unroll
        for (int j = 0; j < 2; ++j) acc[i][j] = (v8f){0,0,0,0,0,0,0,0};

#pragma unroll
    for (int d0 = 0; d0 < DH_; d0 += 32) {
        v16bf a[2], bb[2];
#pragma unroll
        for (int i = 0; i < 2; ++i) {
            const float* row = Qb + (size_t)(qtile + 16 * i + (lane & 15)) * q_stride;
            const int koff = d0 + ((lane & 16) ? 16 : 0);
#pragma unroll
            for (int e = 0; e < 16; ++e)
                a[i][e] = (__bf16)row[koff + e];
        }
#pragma unroll
        for (int j = 0; j < 2; ++j) {
            const float* row = Kb + (size_t)(ktile + 16 * j + (lane & 15)) * k_stride;
            const int koff = d0 + ((lane & 16) ? 16 : 0);
#pragma unroll
            for (int e = 0; e < 16; ++e) bb[j][e] = (__bf16)row[koff + e];
        }
#pragma unroll
        for (int i = 0; i < 2; ++i)
#pragma unroll
            for (int j = 0; j < 2; ++j)
                acc[i][j] = __builtin_amdgcn_wmma_f32_16x16x32_bf16(
                    false, a[i], false, bb[j], (short)0, acc[i][j], false, false);
    }

    const float scale = 0.125f;  // 1/sqrt(64)
#pragma unroll
    for (int i = 0; i < 2; ++i)
#pragma unroll
        for (int j = 0; j < 2; ++j)
#pragma unroll
            for (int v = 0; v < 8; ++v) {
                int sq = qtile + 16 * i + v + ((lane & 16) ? 8 : 0);
                int sk = ktile + 16 * j + (lane & 15);
                float val;
                if (mode == 0) {
                    size_t idx = ((size_t)b * Sq + sq) * Sk + sk;
                    float bv = dist_w[dist[idx] * H_ + h] + iso_w[iso[idx] * H_ + h];
                    if (sk > sq || seqs[b * Sk + sk] == 0) val = -INFINITY;
                    else val = acc[i][j][v] * scale + bv;
                } else {
                    val = memmask[b * Sk + sk] ? -INFINITY : acc[i][j][v] * scale;
                }
                Sout[((size_t)z * Sq + sq) * Sk + sk] = val;
            }
}

// ---------------------------------------------------------------------------
// Row softmax: reads f32 scores, writes bf16 probabilities.
// Row (<=256 cols) held in registers: one global read + one bf16 write.
// ---------------------------------------------------------------------------
__global__ __launch_bounds__(128)
void softmax_bf16_kernel(const float* __restrict__ S, __bf16* __restrict__ P, int cols)
{
    __shared__ float red[128];
    const float* sp = S + (size_t)blockIdx.x * cols;
    __bf16* pp = P + (size_t)blockIdx.x * cols;
    const int t = threadIdx.x;
    const int nit = cols >> 7;       // 1 (cols=128) or 2 (cols=256)
    float v[2];
    float m = -INFINITY;
#pragma unroll 2
    for (int i = 0; i < nit; ++i) { v[i] = sp[t + 128 * i]; m = fmaxf(m, v[i]); }
    red[t] = m; __syncthreads();
    for (int s = 64; s > 0; s >>= 1) { if (t < s) red[t] = fmaxf(red[t], red[t + s]); __syncthreads(); }
    m = red[0]; __syncthreads();
    float sum = 0.0f;
#pragma unroll 2
    for (int i = 0; i < nit; ++i) { v[i] = expf(v[i] - m); sum += v[i]; }
    red[t] = sum; __syncthreads();
    for (int s = 64; s > 0; s >>= 1) { if (t < s) red[t] += red[t + s]; __syncthreads(); }
    float inv = 1.0f / red[0];
#pragma unroll 2
    for (int i = 0; i < nit; ++i) pp[t + 128 * i] = (__bf16)(v[i] * inv);
}

// ---------------------------------------------------------------------------
// context[b, sq, h*DH+d] = sum_k P[z,sq,k] * V_z[k,d]
// P is bf16 (consumed directly as A-fragments, contiguous 32B runs).
// ---------------------------------------------------------------------------
__global__ __launch_bounds__(32)
void attn_av_kernel(const __bf16* __restrict__ P,
                    const float* __restrict__ V, int v_stride,
                    float* __restrict__ O, int Sq, int Sk)
{
    const int lane = threadIdx.x & 31;
    const int z = blockIdx.z;
    const int b = z / H_, h = z % H_;
    const int dtile = blockIdx.x * 32;
    const int qtile = blockIdx.y * 32;
    const __bf16* Pb = P + (size_t)z * Sq * Sk;
    const float* Vb = V + (size_t)b * Sk * v_stride + h * DH_;

    v8f acc[2][2];
#pragma unroll
    for (int i = 0; i < 2; ++i)
#pragma unroll
        for (int j = 0; j < 2; ++j) acc[i][j] = (v8f){0,0,0,0,0,0,0,0};

    for (int k0 = 0; k0 < Sk; k0 += 32) {
        v16bf a[2], bb[2];
#pragma unroll
        for (int i = 0; i < 2; ++i) {
            const __bf16* row = Pb + (size_t)(qtile + 16 * i + (lane & 15)) * Sk;
            const int koff = k0 + ((lane & 16) ? 16 : 0);
#pragma unroll
            for (int e = 0; e < 16; ++e)
                a[i][e] = row[koff + e];
        }
#pragma unroll
        for (int j = 0; j < 2; ++j) {
            const int d = dtile + 16 * j + (lane & 15);
            const int koff = k0 + ((lane & 16) ? 16 : 0);
#pragma unroll
            for (int e = 0; e < 16; ++e)
                bb[j][e] = (__bf16)Vb[(size_t)(koff + e) * v_stride + d];
        }
#pragma unroll
        for (int i = 0; i < 2; ++i)
#pragma unroll
            for (int j = 0; j < 2; ++j)
                acc[i][j] = __builtin_amdgcn_wmma_f32_16x16x32_bf16(
                    false, a[i], false, bb[j], (short)0, acc[i][j], false, false);
    }
#pragma unroll
    for (int i = 0; i < 2; ++i)
#pragma unroll
        for (int j = 0; j < 2; ++j)
#pragma unroll
            for (int v = 0; v < 8; ++v) {
                int sq = qtile + 16 * i + v + ((lane & 16) ? 8 : 0);
                int d  = dtile + 16 * j + (lane & 15);
                O[((size_t)b * Sq + sq) * E_ + h * DH_ + d] = acc[i][j][v];
            }
}

// ---------------------------------------------------------------------------
// x = LN(x + res) * g + b  (in place; res may be null). One block per row, E=1024.
// ---------------------------------------------------------------------------
__global__ __launch_bounds__(256)
void ln_kernel(float* __restrict__ x, const float* __restrict__ res,
               const float* __restrict__ g, const float* __restrict__ bb)
{
    __shared__ float r1[256], r2[256];
    float* xp = x + (size_t)blockIdx.x * E_;
    const float* rp = res ? res + (size_t)blockIdx.x * E_ : nullptr;
    const int t = threadIdx.x;
    float v[4], s = 0.0f, s2 = 0.0f;
#pragma unroll
    for (int i = 0; i < 4; ++i) {
        int e = t + 256 * i;
        float val = xp[e] + (rp ? rp[e] : 0.0f);
        v[i] = val; s += val; s2 += val * val;
    }
    r1[t] = s; r2[t] = s2; __syncthreads();
    for (int k = 128; k > 0; k >>= 1) {
        if (t < k) { r1[t] += r1[t + k]; r2[t] += r2[t + k]; }
        __syncthreads();
    }
    float mean = r1[0] * (1.0f / E_);
    float var  = r2[0] * (1.0f / E_) - mean * mean;
    float inv  = rsqrtf(var + 1e-5f);
#pragma unroll
    for (int i = 0; i < 4; ++i) {
        int e = t + 256 * i;
        xp[e] = (v[i] - mean) * inv * g[e] + bb[e];
    }
}

// ---------------------------------------------------------------------------
// x[bs,e] = tok_w[seq[bs],e]*sqrt(E) + posenc(s,e). One block per token.
// ---------------------------------------------------------------------------
__global__ __launch_bounds__(256)
void embed_kernel(const int* __restrict__ seqs, const float* __restrict__ tok_w,
                  float* __restrict__ x)
{
    const size_t bs = blockIdx.x;
    const int s = (int)(bs % S_);
    const int tok = seqs[bs];
    const float sqE = 32.0f;  // sqrt(1024)
    for (int e = threadIdx.x; e < E_; e += 256) {
        int e2 = e & ~1;
        float den = expf(-(float)e2 * (9.210340371976184f / (float)E_));  // ln(1e4)
        float ang = (float)s * den;
        float pe = (e & 1) ? cosf(ang) : sinf(ang);
        x[bs * E_ + e] = tok_w[(size_t)tok * E_ + e] * sqE + pe;
    }
}

// ---------------------------------------------------------------------------
extern "C" void kernel_launch(void* const* d_in, const int* in_sizes, int n_in,
                              void* d_out, int out_size, void* d_ws, size_t ws_size,
                              hipStream_t stream)
{
    const int*   seqs    = (const int*)  d_in[0];
    const int*   dist    = (const int*)  d_in[1];
    const int*   iso     = (const int*)  d_in[2];
    const float* memory  = (const float*)d_in[3];
    const unsigned char* memmask = (const unsigned char*)d_in[4];
    const float* tok_w   = (const float*)d_in[5];
    const float* dist_w  = (const float*)d_in[6];
    const float* iso_w   = (const float*)d_in[7];
    const float* Wqkv_s  = (const float*)d_in[8];
    const float* bqkv_s  = (const float*)d_in[9];
    const float* Wo_s    = (const float*)d_in[10];
    const float* bo_s    = (const float*)d_in[11];
    const float* Wqkv_c  = (const float*)d_in[12];
    const float* bqkv_c  = (const float*)d_in[13];
    const float* Wo_c    = (const float*)d_in[14];
    const float* bo_c    = (const float*)d_in[15];
    const float* W1      = (const float*)d_in[16];
    const float* b1      = (const float*)d_in[17];
    const float* W2      = (const float*)d_in[18];
    const float* b2      = (const float*)d_in[19];
    const float* ln1_g   = (const float*)d_in[20];
    const float* ln1_b   = (const float*)d_in[21];
    const float* ln2_g   = (const float*)d_in[22];
    const float* ln2_b   = (const float*)d_in[23];
    const float* ln3_g   = (const float*)d_in[24];
    const float* ln3_b   = (const float*)d_in[25];
    const float* lnf_g   = (const float*)d_in[26];
    const float* lnf_b   = (const float*)d_in[27];
    const float* gen_w   = (const float*)d_in[28];
    const float* gen_b   = (const float*)d_in[29];
    float* out = (float*)d_out;

    // workspace carve-out
    char* ws = (char*)d_ws;
    size_t off = 0;
    auto carveb = [&](size_t bytes) { void* p = ws + off; off += (bytes + 255) & ~(size_t)255; return p; };
    float*  xf     = (float*) carveb((size_t)B_ * S_ * E_ * 4);        // activations
    float*  qkv    = (float*) carveb((size_t)B_ * S_ * 3 * E_ * 4);    // qkv / q proj
    float*  scores = (float*) carveb((size_t)B_ * H_ * S_ * S_ * 4);   // raw scores
    __bf16* probs  = (__bf16*)carveb((size_t)B_ * H_ * S_ * S_ * 2);   // softmax out
    float*  ctx    = (float*) carveb((size_t)B_ * S_ * E_ * 4);        // attn context
    float*  proj   = (float*) carveb((size_t)B_ * S_ * E_ * 4);        // residual branch
    float*  ffh    = (float*) carveb((size_t)B_ * S_ * F_ * 4);        // FFN hidden
    float*  kvmem  = (float*) carveb((size_t)B_ * M_ * 2 * E_ * 4);    // cross K,V
    (void)ws_size; (void)in_sizes; (void)n_in; (void)out_size;

    const int BS = B_ * S_;   // 8192
    const int BM = B_ * M_;   // 4096
    dim3 blk256(256), blk128(128), blk32(32);

    embed_kernel<<<BS, blk256, 0, stream>>>(seqs, tok_w, xf);

    for (int l = 0; l < L_; ++l) {
        // ---- self attention ----
        gemm_bf16_kernel<<<dim3((3 * E_ + TN - 1) / TN, BS / TM), blk256, 0, stream>>>(
            xf, Wqkv_s + (size_t)l * 3 * E_ * E_, bqkv_s + (size_t)l * 3 * E_,
            qkv, BS, 3 * E_, E_, 0);
        attn_scores_kernel<<<dim3(S_ / 32, S_ / 32, B_ * H_), blk32, 0, stream>>>(
            qkv, 3 * E_, qkv + E_, 3 * E_, scores, S_, S_,
            dist, iso, dist_w, iso_w, seqs, nullptr, 0);
        softmax_bf16_kernel<<<(unsigned)(B_ * H_ * S_), blk128, 0, stream>>>(scores, probs, S_);
        attn_av_kernel<<<dim3(DH_ / 32, S_ / 32, B_ * H_), blk32, 0, stream>>>(
            probs, qkv + 2 * E_, 3 * E_, ctx, S_, S_);
        gemm_bf16_kernel<<<dim3((E_ + TN - 1) / TN, BS / TM), blk256, 0, stream>>>(
            ctx, Wo_s + (size_t)l * E_ * E_, bo_s + (size_t)l * E_, proj, BS, E_, E_, 0);
        ln_kernel<<<BS, blk256, 0, stream>>>(xf, proj, ln1_g + (size_t)l * E_, ln1_b + (size_t)l * E_);

        // ---- cross attention ----
        gemm_bf16_kernel<<<dim3((E_ + TN - 1) / TN, BS / TM), blk256, 0, stream>>>(
            xf, Wqkv_c + (size_t)l * 3 * E_ * E_, bqkv_c + (size_t)l * 3 * E_,
            qkv, BS, E_, E_, 0);  // Q only, compact [BS,E]
        gemm_bf16_kernel<<<dim3((2 * E_ + TN - 1) / TN, BM / TM), blk256, 0, stream>>>(
            memory, Wqkv_c + (size_t)l * 3 * E_ * E_ + (size_t)E_ * E_,
            bqkv_c + (size_t)l * 3 * E_ + E_, kvmem, BM, 2 * E_, E_, 0);  // K,V
        attn_scores_kernel<<<dim3(M_ / 32, S_ / 32, B_ * H_), blk32, 0, stream>>>(
            qkv, E_, kvmem, 2 * E_, scores, S_, M_,
            nullptr, nullptr, nullptr, nullptr, nullptr, memmask, 1);
        softmax_bf16_kernel<<<(unsigned)(B_ * H_ * S_), blk128, 0, stream>>>(scores, probs, M_);
        attn_av_kernel<<<dim3(DH_ / 32, S_ / 32, B_ * H_), blk32, 0, stream>>>(
            probs, kvmem + E_, 2 * E_, ctx, S_, M_);
        gemm_bf16_kernel<<<dim3((E_ + TN - 1) / TN, BS / TM), blk256, 0, stream>>>(
            ctx, Wo_c + (size_t)l * E_ * E_, bo_c + (size_t)l * E_, proj, BS, E_, E_, 0);
        ln_kernel<<<BS, blk256, 0, stream>>>(xf, proj, ln2_g + (size_t)l * E_, ln2_b + (size_t)l * E_);

        // ---- FFN ----
        gemm_bf16_kernel<<<dim3((F_ + TN - 1) / TN, BS / TM), blk256, 0, stream>>>(
            xf, W1 + (size_t)l * F_ * E_, b1 + (size_t)l * F_, ffh, BS, F_, E_, 1);
        gemm_bf16_kernel<<<dim3((E_ + TN - 1) / TN, BS / TM), blk256, 0, stream>>>(
            ffh, W2 + (size_t)l * E_ * F_, b2 + (size_t)l * E_, proj, BS, E_, F_, 0);
        ln_kernel<<<BS, blk256, 0, stream>>>(xf, proj, ln3_g + (size_t)l * E_, ln3_b + (size_t)l * E_);
    }

    ln_kernel<<<BS, blk256, 0, stream>>>(xf, nullptr, lnf_g, lnf_b);
    gemm_bf16_kernel<<<dim3((V_ + TN - 1) / TN, BS / TM), blk256, 0, stream>>>(
        xf, gen_w, gen_b, out, BS, V_, E_, 0);
}
